// IndRNNCell_25357486916093
// MI455X (gfx1250) — matrix operand order repaired and checked by reading, not compiled, not run on hardware
//
#include <hip/hip_runtime.h>

// CDNA5 (gfx1250) wave32 WMMA types
typedef __attribute__((ext_vector_type(16))) __bf16 v16bf;
typedef __attribute__((ext_vector_type(8)))  float  v8f;

#define I_SIZE 256
#define H_SIZE 256
#define BATCH  65536
// LDS row stride in bf16 elements: 256 + 8 pad -> 528 bytes = 132 dwords/row.
// 132 mod 64 = 4, so lane n's ds_load_b128 hits banks 4n+{0..3}: conflict-free.
#define LDS_STRIDE 264

__device__ __forceinline__ unsigned int f2bf_rne(float f) {
    unsigned int u = __float_as_uint(f);
    unsigned int r = u + 0x7FFFu + ((u >> 16) & 1u);   // round-to-nearest-even
    return r >> 16;
}
__device__ __forceinline__ unsigned int pack2bf(float lo, float hi) {
    return f2bf_rne(lo) | (f2bf_rne(hi) << 16);
}

__global__ __launch_bounds__(256) void indrnn_wmma_kernel(
    const float* __restrict__ x,       // [B, I]
    const float* __restrict__ h_prev,  // [B, H]
    const float* __restrict__ W,       // [H, I] row-major
    const float* __restrict__ b_lin,   // [H]
    const float* __restrict__ u,       // [H]
    const float* __restrict__ bias,    // [H]
    float* __restrict__ out)           // [B, H]
{
    extern __shared__ __bf16 ldsW[];   // [H_SIZE][LDS_STRIDE] bf16 = 132 KB

    const int tid = threadIdx.x;

    // ---- Stage W into LDS as bf16 (cooperative, vectorized: b128 in, b64 out)
    // 16384 chunks of 4 floats; 64 chunks per row.
    for (int c = tid; c < (H_SIZE * I_SIZE) / 4; c += 256) {
        const int row = c >> 6;            // I_SIZE/4 = 64 chunks per row
        const int col = (c & 63) << 2;     // element column
        float4 v = ((const float4*)W)[c];
        unsigned int p0 = pack2bf(v.x, v.y);
        unsigned int p1 = pack2bf(v.z, v.w);
        unsigned int* dst = (unsigned int*)(ldsW + row * LDS_STRIDE + col);
        dst[0] = p0;
        dst[1] = p1;
    }
    __syncthreads();

    // ---- Per-wave tile setup: wave w owns batch rows [rowBase, rowBase+16)
    const int wave = tid >> 5;
    const int lane = tid & 31;
    const int rowBase = blockIdx.x * 128 + wave * 16;

    // ---- Load A fragments (x tile, 16x256) into registers as bf16.
    // ISA 16-bit A 16x32 layout: lanes 0-15 hold M=lane, K-halves 0..7 / 16..23;
    // lanes 16-31 hold M=lane-16, K-halves 8..15 / 24..31.
    const int m_lane = lane & 15;
    const int khalf  = (lane >> 4) * 8;
    const float* xrow = x + (size_t)(rowBase + m_lane) * I_SIZE;

    v16bf afrag[8];
    #pragma unroll
    for (int ks = 0; ks < 8; ++ks) {
        const int k0 = ks * 32 + khalf;        // K for VGPRs 0..3
        const int k1 = ks * 32 + 16 + khalf;   // K for VGPRs 4..7
        float4 a0 = *(const float4*)(xrow + k0);
        float4 a1 = *(const float4*)(xrow + k0 + 4);
        float4 a2 = *(const float4*)(xrow + k1);
        float4 a3 = *(const float4*)(xrow + k1 + 4);
        union { v16bf v; unsigned int w[8]; } fr;
        fr.w[0] = pack2bf(a0.x, a0.y);
        fr.w[1] = pack2bf(a0.z, a0.w);
        fr.w[2] = pack2bf(a1.x, a1.y);
        fr.w[3] = pack2bf(a1.z, a1.w);
        fr.w[4] = pack2bf(a2.x, a2.y);
        fr.w[5] = pack2bf(a2.z, a2.w);
        fr.w[6] = pack2bf(a3.x, a3.y);
        fr.w[7] = pack2bf(a3.z, a3.w);
        afrag[ks] = fr.v;
    }

    // ---- GEMM over 16 N-tiles of H; K=256 = 8 x (K=32) WMMA steps each.
    // ISA 16-bit B 32x16 layout: lanes 0-15 are columns N with K rows 0..7 / 16..23
    // (2 per VGPR); lanes 16-31 are the same columns with K rows 8..15 / 24..31.
    const int n_lane = lane & 15;
    const int kb_off = (lane >> 4) * 8;        // B K-row offset within K=32 step
    const int m0     = rowBase + (lane >> 4) * 8; // C/D layout: M = m0 + r

    for (int nt = 0; nt < 16; ++nt) {
        const int h = nt * 16 + n_lane;
        const __bf16* wrow = ldsW + h * LDS_STRIDE;

        v8f acc = {};
        #pragma unroll
        for (int ks = 0; ks < 8; ++ks) {
            // Two 16B chunks from W row h (K kb_off..+7 and 16+kb_off..+7), alignment 16.
            union { v16bf v; uint4 q[2]; } bf;
            bf.q[0] = *(const uint4*)(wrow + ks * 32 + kb_off);
            bf.q[1] = *(const uint4*)(wrow + ks * 32 + kb_off + 16);
            acc = __builtin_amdgcn_wmma_f32_16x16x32_bf16(
                /*neg_a=*/false, afrag[ks],
                /*neg_b=*/false, bf.v,
                /*c_mod=*/(short)0, acc,
                /*reuse_a=*/false, /*reuse_b=*/false);
        }

        // ---- Fused epilogue: + (b_lin + bias) + u * h_prev, then ReLU.
        const float bl = b_lin[h] + bias[h];
        const float uu = u[h];
        #pragma unroll
        for (int r = 0; r < 8; ++r) {
            const size_t idx = (size_t)(m0 + r) * H_SIZE + h;
            float v = acc[r] + bl + uu * h_prev[idx];
            out[idx] = fmaxf(v, 0.0f);
        }
    }
}

extern "C" void kernel_launch(void* const* d_in, const int* in_sizes, int n_in,
                              void* d_out, int out_size, void* d_ws, size_t ws_size,
                              hipStream_t stream) {
    (void)in_sizes; (void)n_in; (void)out_size; (void)d_ws; (void)ws_size;
    const float* x      = (const float*)d_in[0];
    const float* h_prev = (const float*)d_in[1];
    const float* W      = (const float*)d_in[2];
    const float* b_lin  = (const float*)d_in[3];
    const float* u      = (const float*)d_in[4];
    const float* bias   = (const float*)d_in[5];
    float* out = (float*)d_out;

    const size_t lds_bytes = (size_t)H_SIZE * LDS_STRIDE * sizeof(__bf16); // 132 KB
    dim3 grid(BATCH / 128);   // 512 blocks; each block = 8 waves x 16 batch rows
    dim3 block(256);
    indrnn_wmma_kernel<<<grid, block, lds_bytes, stream>>>(
        x, h_prev, W, b_lin, u, bias, out);
}
